// VisionTransformerMoE_53910429499609
// MI455X (gfx1250) — matrix operand order repaired and checked by reading, not compiled, not run on hardware
//
#include <hip/hip_runtime.h>
#include <hip/hip_bf16.h>

// ---------------------------------------------------------------------------
// CDNA5 (gfx1250) ViT-MoE forward.
// All GEMMs run through one wave32 WMMA kernel (v_wmma_f32_16x16x32_bf16).
// Model is HBM-bandwidth bound (~115 GFLOP, ~284MB fp32 weights), so weights
// and activations are carried in bf16 (halves bytes, fast dense 16-bit WMMA
// path with f32 accumulation), and every epilogue is fused (bias, exact-erf
// GELU, residual +=, MoE combine scale, attention alpha, bf16 requant).
// Tile loads: global_load_b128 -> ds_store_b128 fast path for interior tiles,
// branchless clamped loads for edges (no per-element exec branching).
// ---------------------------------------------------------------------------

typedef __attribute__((ext_vector_type(16))) __bf16 v16bf;
typedef __attribute__((ext_vector_type(8)))  __bf16 v8bf;   // 16 bytes
typedef __attribute__((ext_vector_type(8)))  float  v8f;

#define TM  64
#define TN  128
#define TKK 32

// Generic tiled GEMM:  C[m,n] = alpha * sum_k A[m,k] * B(k,n)  (+bias[n])
//   A  : bf16, row-major, leading dim lda, optional z-batch offsets
//   Bw : bf16; if b_rowmajor==0 it is W[N,K] row-major (C = A*W^T),
//        if b_rowmajor==1 it is B[K,N] row-major.
// Block = 256 threads = 8 wave32, tile 64x128; each wave computes a 32x32
// sub-tile: 4 accumulators, A-fragments reused across 2 B-fragments ->
// 4 x v_wmma_f32_16x16x32_bf16 per wave per K-step (K-step 32).
__global__ __launch_bounds__(256) void k_gemm_bf16(
    const __bf16* __restrict__ A, long lda, long sA1, long sA2,
    const __bf16* __restrict__ Bw, long ldb, long sB1, long sB2, int b_rowmajor,
    const float* __restrict__ bias,
    float* __restrict__ Of, long ldo, long sO1, long sO2, int accum,
    __bf16* __restrict__ Ob, long ldob, long sOb1, long sOb2,
    const float* __restrict__ rowscale, int rs_stride,
    int M, int N, int K, int z2n, int fuse_gelu, float alpha)
{
  int z  = blockIdx.z;
  int z1 = z / z2n, z2 = z - z1 * z2n;
  A  += (long)z1 * sA1 + (long)z2 * sA2;
  Bw += (long)z1 * sB1 + (long)z2 * sB2;
  long oOff  = (long)z1 * sO1  + (long)z2 * sO2;
  long obOff = (long)z1 * sOb1 + (long)z2 * sOb2;

  __shared__ __align__(16) __bf16 lA[TM][TKK];   // [m][k]
  __shared__ __align__(16) __bf16 lB[TN][TKK];   // [n][k]  (transposed stage)

  int tid  = threadIdx.x;
  int wave = tid >> 5, lane = tid & 31;
  int wm = wave & 1, wn = wave >> 1;        // 2 (M) x 4 (N) waves
  int r16 = lane & 15, hh = lane >> 4;      // wave32 lane halves
  int m0 = blockIdx.y * TM, n0 = blockIdx.x * TN;

  v8f acc[4] = {};                          // [s*2+cg]

  for (int k0 = 0; k0 < K; k0 += TKK) {
    __syncthreads();

    // ---- stage A tile (64x32 bf16) ----
    if (m0 + TM <= M && k0 + TKK <= K) {
      // interior: one 16B chunk per thread (global_load_b128 / ds_store_b128)
      int r = tid >> 2, ch = tid & 3;
      const __bf16* src = A + (long)(m0 + r) * lda + k0 + ch * 8;
      *(v8bf*)&lA[r][ch * 8] = *(const v8bf*)src;
      if (k0 + TKK < K) __builtin_prefetch(src + TKK, 0, 1);
    } else {
      // edge: branchless clamped loads (v_cndmask, no exec branching)
      #pragma unroll 2
      for (int i = tid; i < TM * TKK; i += 256) {
        int r = i >> 5, c = i & 31;
        int gm = m0 + r, gk = k0 + c;
        int sm = gm < M ? gm : M - 1;
        int sk = gk < K ? gk : K - 1;
        __bf16 v = A[(long)sm * lda + sk];
        if (gm >= M || gk >= K) v = (__bf16)0.f;
        lA[r][c] = v;
      }
    }

    // ---- stage B tile (128x32 bf16, [n][k] layout) ----
    if (!b_rowmajor && n0 + TN <= N && k0 + TKK <= K) {
      // interior W-layout: rows of W map directly to LDS rows, 2x16B per thread
      #pragma unroll
      for (int i = tid; i < (TN * TKK) / 8; i += 256) {
        int r = i >> 2, ch = i & 3;
        const __bf16* src = Bw + (long)(n0 + r) * ldb + k0 + ch * 8;
        *(v8bf*)&lB[r][ch * 8] = *(const v8bf*)src;
      }
    } else {
      #pragma unroll 2
      for (int i = tid; i < TN * TKK; i += 256) {
        int r = i >> 5, c = i & 31;
        int gn = n0 + r, gk = k0 + c;
        int sn = gn < N ? gn : N - 1;
        int sk = gk < K ? gk : K - 1;
        __bf16 v = b_rowmajor ? Bw[(long)sk * ldb + sn]
                              : Bw[(long)sn * ldb + sk];
        if (gn >= N || gk >= K) v = (__bf16)0.f;
        lB[r][c] = v;
      }
    }
    __syncthreads();

    // B fragments (32x16 bf16): lane holds column n; VGPR j holds
    // K = hh*16 + 2j, 2j+1 -> contiguous pairs in the lB row.
    union { v16bf v; unsigned u[8]; } fb[2];
    #pragma unroll
    for (int cg = 0; cg < 2; ++cg) {
      const unsigned* row = (const unsigned*)&lB[wn * 32 + cg * 16 + r16][0];
      #pragma unroll
      for (int j = 0; j < 8; ++j) fb[cg].u[j] = row[hh * 8 + j];
    }
    #pragma unroll
    for (int s = 0; s < 2; ++s) {
      // A fragment (16x32 bf16): lane holds row r16; VGPR j holds
      // K = hh*16 + 2j, 2j+1 (same order as B)
      union { v16bf v; unsigned u[8]; } fa;
      const unsigned* row = (const unsigned*)&lA[wm * 32 + s * 16 + r16][0];
      #pragma unroll
      for (int j = 0; j < 8; ++j) {
        int kp = hh * 8 + j;
        fa.u[j] = row[kp];
      }
      #pragma unroll
      for (int cg = 0; cg < 2; ++cg) {
        acc[s * 2 + cg] = __builtin_amdgcn_wmma_f32_16x16x32_bf16(
            false, fa.v, false, fb[cg].v, (short)0, acc[s * 2 + cg],
            false, false);
      }
    }
  }

  // epilogue: C/D layout -> lane L: N = L&15 (both halves), VGPR j: M = j+8*(L>>4)
  #pragma unroll
  for (int cg = 0; cg < 2; ++cg) {
    int nCol = n0 + wn * 32 + cg * 16 + r16;
    if (nCol >= N) continue;
    float bval = bias ? bias[nCol] : 0.f;
    #pragma unroll
    for (int s = 0; s < 2; ++s) {
      #pragma unroll
      for (int j = 0; j < 8; ++j) {
        int gm = m0 + wm * 32 + s * 16 + j + 8 * hh;
        if (gm < M) {
          float val = alpha * acc[s * 2 + cg][j] + bval;
          if (fuse_gelu) val = 0.5f * val * (1.f + erff(val * 0.70710678118f));
          if (rowscale)  val *= rowscale[(long)gm * rs_stride];
          if (Of) {
            long o = oOff + (long)gm * ldo + nCol;
            if (accum) Of[o] += val; else Of[o] = val;
          }
          if (Ob) Ob[obOff + (long)gm * ldob + nCol] = (__bf16)val;
        }
      }
    }
  }
}

// LayerNorm (biased variance, eps inside sqrt — matches reference),
// writes bf16 (GEMM operand) and optionally fp32 (MoE gating input).
__global__ __launch_bounds__(128) void k_layernorm(
    const float* __restrict__ x, long rstride,
    const float* __restrict__ w, const float* __restrict__ b,
    __bf16* __restrict__ yb, long ldyb,
    float* __restrict__ yf, long ldyf, int rows, int D)
{
  int row = blockIdx.x; if (row >= rows) return;
  const float* xr = x + (long)row * rstride;
  __shared__ float shs[128], shq[128];
  int tid = threadIdx.x;
  float s = 0.f, q = 0.f;
  for (int d = tid; d < D; d += 128) { float v = xr[d]; s += v; q += v * v; }
  shs[tid] = s; shq[tid] = q; __syncthreads();
  for (int off = 64; off > 0; off >>= 1) {
    if (tid < off) { shs[tid] += shs[tid + off]; shq[tid] += shq[tid + off]; }
    __syncthreads();
  }
  float mean = shs[0] / D;
  float var  = shq[0] / D - mean * mean;
  float inv  = rsqrtf(var + 1e-6f);
  for (int d = tid; d < D; d += 128) {
    float v = (xr[d] - mean) * inv * w[d] + b[d];
    if (yb) yb[(long)row * ldyb + d] = (__bf16)v;
    if (yf) yf[(long)row * ldyf + d] = v;
  }
}

// Row softmax over fp32 scores -> bf16 probabilities (padded cols zeroed).
__global__ __launch_bounds__(256) void k_softmax_rows(
    const float* __restrict__ S, __bf16* __restrict__ P,
    int rows, int cols, int ldS, int ldP)
{
  long base = (long)blockIdx.y * rows + blockIdx.x;
  const float* sr = S + base * ldS;
  __bf16* pr = P + base * ldP;
  __shared__ float sh[256];
  int tid = threadIdx.x;
  float m = -3.4e38f;
  for (int c = tid; c < cols; c += 256) m = fmaxf(m, sr[c]);
  sh[tid] = m; __syncthreads();
  for (int off = 128; off > 0; off >>= 1) { if (tid < off) sh[tid] = fmaxf(sh[tid], sh[tid + off]); __syncthreads(); }
  m = sh[0]; __syncthreads();
  float s = 0.f;
  for (int c = tid; c < cols; c += 256) s += expf(sr[c] - m);
  sh[tid] = s; __syncthreads();
  for (int off = 128; off > 0; off >>= 1) { if (tid < off) sh[tid] += sh[tid + off]; __syncthreads(); }
  float inv = 1.f / sh[0];
  for (int c = tid; c < ldP; c += 256)
    pr[c] = (c < cols) ? (__bf16)(expf(sr[c] - m) * inv) : (__bf16)0.f;
}

// MoE top-2 gating: 8 wave32, one expert per wave (lane-strided dot + shuffle
// reduce), then renormalized softmax over the top-2 -> combine[T,8].
__global__ __launch_bounds__(256) void k_moe_gate(
    const float* __restrict__ t, const float* __restrict__ gw,
    float* __restrict__ combine, int T, int D, int E)
{
  int tok = blockIdx.x; if (tok >= T) return;
  int wave = threadIdx.x >> 5, lane = threadIdx.x & 31;
  __shared__ float lg[8];
  if (wave < E) {
    const float* x = t  + (long)tok  * D;
    const float* g = gw + (long)wave * D;
    float s = 0.f;
    for (int d = lane; d < D; d += 32) s += x[d] * g[d];
    for (int off = 16; off > 0; off >>= 1) s += __shfl_down(s, off, 32);
    if (lane == 0) lg[wave] = s;
  }
  __syncthreads();
  if (threadIdx.x == 0) {
    int i0 = 0;
    for (int e = 1; e < E; ++e) if (lg[e] > lg[i0]) i0 = e;
    int i1 = -1;
    for (int e = 0; e < E; ++e) { if (e == i0) continue; if (i1 < 0 || lg[e] > lg[i1]) i1 = e; }
    float m  = lg[i0];
    float e0 = expf(lg[i0] - m), e1 = expf(lg[i1] - m);
    float inv = 1.f / (e0 + e1);
    for (int e = 0; e < E; ++e) combine[(long)tok * E + e] = 0.f;
    combine[(long)tok * E + i0] = e0 * inv;
    combine[(long)tok * E + i1] = e1 * inv;
  }
}

// im2col for the stride-16 patch conv: token row = 768 bf16 (c*256 + i*16 + j)
__global__ __launch_bounds__(256) void k_im2col(
    const float* __restrict__ x, __bf16* __restrict__ out)
{
  int tkn = blockIdx.x;                 // b*196 + p
  int b = tkn / 196, p = tkn % 196;
  int pi = p / 14, pj = p % 14;
  for (int idx = threadIdx.x; idx < 768; idx += 256) {
    int c = idx >> 8, r = (idx >> 4) & 15, col = idx & 15;
    float v = x[(((long)b * 3 + c) * 224 + (pi * 16 + r)) * 224 + (pj * 16 + col)];
    out[(long)tkn * 768 + idx] = (__bf16)v;
  }
}

// h[b,n,:] = (n==0 ? cls : tok[b*196+n-1]) + pos[n]
__global__ __launch_bounds__(128) void k_assemble(
    const float* __restrict__ tok, const float* __restrict__ cls,
    const float* __restrict__ pos, float* __restrict__ h)
{
  int rn = blockIdx.x;                  // b*197 + n
  int b = rn / 197, n = rn % 197;
  float* hr = h + (long)rn * 384;
  for (int d = threadIdx.x; d < 384; d += 128) {
    float v = (n == 0) ? cls[d] : tok[((long)b * 196 + (n - 1)) * 384 + d];
    hr[d] = v + pos[(long)n * 384 + d];
  }
}

__global__ __launch_bounds__(256) void k_cvt_bf16(
    const float* __restrict__ s, __bf16* __restrict__ d, long n)
{
  long i  = (long)blockIdx.x * blockDim.x + threadIdx.x;
  long st = (long)gridDim.x * blockDim.x;
  for (; i < n; i += st) d[i] = (__bf16)s[i];
}

// ---------------------------------------------------------------------------
static inline void launch_gemm(hipStream_t stream,
    const __bf16* A, long lda, long sA1, long sA2,
    const __bf16* Bw, long ldb, long sB1, long sB2, int browm,
    const float* bias,
    float* Of, long ldo, long sO1, long sO2, int accum,
    __bf16* Ob, long ldob, long sOb1, long sOb2,
    const float* rs, int rss,
    int M, int N, int K, int nz, int z2n, int gelu, float alpha)
{
  dim3 g((N + TN - 1) / TN, (M + TM - 1) / TM, nz);
  k_gemm_bf16<<<g, 256, 0, stream>>>(A, lda, sA1, sA2, Bw, ldb, sB1, sB2, browm,
      bias, Of, ldo, sO1, sO2, accum, Ob, ldob, sOb1, sOb2,
      rs, rss, M, N, K, z2n, gelu, alpha);
}

extern "C" void kernel_launch(void* const* d_in, const int* in_sizes, int n_in,
                              void* d_out, int out_size, void* d_ws, size_t ws_size,
                              hipStream_t stream)
{
  (void)in_sizes; (void)n_in; (void)out_size; (void)ws_size;
  constexpr int T = 788, SP = 224;           // tokens, padded attn width (>=197)
  const float* x        = (const float*)d_in[0];
  const float* patch_w  = (const float*)d_in[1];
  const float* patch_b  = (const float*)d_in[2];
  const float* cls_tok  = (const float*)d_in[3];
  const float* pos_emb  = (const float*)d_in[4];
  const float* ln1_w    = (const float*)d_in[5];
  const float* ln1_b    = (const float*)d_in[6];
  const float* qkv_w    = (const float*)d_in[7];
  const float* qkv_b    = (const float*)d_in[8];
  const float* proj_w   = (const float*)d_in[9];
  const float* proj_b   = (const float*)d_in[10];
  const float* ln2_w    = (const float*)d_in[11];
  const float* ln2_b    = (const float*)d_in[12];
  const float* fc1_w    = (const float*)d_in[13];
  const float* fc1_b    = (const float*)d_in[14];
  const float* fc2_w    = (const float*)d_in[15];
  const float* fc2_b    = (const float*)d_in[16];
  const float* gate_w   = (const float*)d_in[17];
  const float* e1_w     = (const float*)d_in[18];
  const float* e1_b     = (const float*)d_in[19];
  const float* e2_w     = (const float*)d_in[20];
  const float* e2_b     = (const float*)d_in[21];
  const float* lnf_w    = (const float*)d_in[22];
  const float* lnf_b    = (const float*)d_in[23];
  const float* head_w   = (const float*)d_in[24];
  const float* head_b   = (const float*)d_in[25];
  float* out = (float*)d_out;

  // workspace carve (~20 MB)
  char* wp = (char*)d_ws;
  auto carve = [&](size_t bytes) -> void* {
    void* p = wp; wp += (bytes + 255) & ~(size_t)255; return p;
  };
  __bf16* wbuf  = (__bf16*)carve((size_t)1536 * 384 * 2);  // per-layer weight stage
  __bf16* im2c  = (__bf16*)carve((size_t)784 * 768 * 2);
  float*  tokf  = (float*) carve((size_t)784 * 384 * 4);
  float*  h     = (float*) carve((size_t)T * 384 * 4);
  __bf16* lnb   = (__bf16*)carve((size_t)T * 384 * 2);
  float*  lnf32 = (float*) carve((size_t)T * 384 * 4);
  __bf16* qkvb  = (__bf16*)carve((size_t)T * 1152 * 2);
  float*  Sb    = (float*) carve((size_t)24 * 197 * SP * 4);
  __bf16* Pb    = (__bf16*)carve((size_t)24 * 197 * SP * 2);
  __bf16* attb  = (__bf16*)carve((size_t)T * 384 * 2);
  __bf16* ffnb  = (__bf16*)carve((size_t)T * 1536 * 2);
  float*  comb  = (float*) carve((size_t)T * 8 * 4);
  __bf16* clsb  = (__bf16*)carve((size_t)4 * 384 * 2);

  auto cvt = [&](const float* s, __bf16* d, long n) {
    int blocks = (int)((n + 255) / 256); if (blocks > 2048) blocks = 2048;
    k_cvt_bf16<<<blocks, 256, 0, stream>>>(s, d, n);
  };

  // patch embed = im2col + GEMM (W = patch_w [384,768] row-major, bias fused)
  k_im2col<<<784, 256, 0, stream>>>(x, im2c);
  cvt(patch_w, wbuf, 384L * 768);
  launch_gemm(stream, im2c, 768, 0, 0, wbuf, 768, 0, 0, 0, patch_b,
              tokf, 384, 0, 0, 0, nullptr, 0, 0, 0, nullptr, 0,
              784, 384, 768, 1, 1, 0, 1.f);
  k_assemble<<<T, 128, 0, stream>>>(tokf, cls_tok, pos_emb, h);

  const long zA = 197L * 1152;   // per-batch stride inside qkv buffer
  const long zS = 197L * SP;     // per-(b,h) score tile

  for (int l = 0; l < 12; ++l) {
    int m = l / 2;
    // LN1 -> bf16
    k_layernorm<<<T, 128, 0, stream>>>(h, 384, ln1_w + l * 384, ln1_b + l * 384,
                                       lnb, 384, nullptr, 0, T, 384);
    // qkv: [T,384] x [1152,384]^T -> bf16 [T,1152]
    cvt(qkv_w + (long)l * 1152 * 384, wbuf, 1152L * 384);
    launch_gemm(stream, lnb, 384, 0, 0, wbuf, 384, 0, 0, 0, qkv_b + l * 1152,
                nullptr, 0, 0, 0, 0, qkvb, 1152, 0, 0, nullptr, 0,
                T, 1152, 384, 1, 1, 0, 1.f);
    // scores: per (b,head): S = q @ k^T * HD^-0.5   (z = 24, z2n = 6 heads)
    launch_gemm(stream, qkvb, 1152, zA, 64, qkvb + 384, 1152, zA, 64, 0,
                nullptr, Sb, SP, 6 * zS, zS, 0,
                nullptr, 0, 0, 0, nullptr, 0,
                197, 197, 64, 24, 6, 0, 0.125f);
    k_softmax_rows<<<dim3(197, 24), 256, 0, stream>>>(Sb, Pb, 197, 197, SP, SP);
    // out: P @ V  (V row-major [197,64])  -> bf16 head-concat [T,384]
    launch_gemm(stream, Pb, SP, 6 * zS, zS, qkvb + 768, 1152, zA, 64, 1,
                nullptr, nullptr, 0, 0, 0, 0,
                attb, 384, 197L * 384, 64, nullptr, 0,
                197, 64, 197, 24, 6, 0, 1.f);
    // proj: h += attb @ proj_w^T + proj_b
    cvt(proj_w + (long)l * 384 * 384, wbuf, 384L * 384);
    launch_gemm(stream, attb, 384, 0, 0, wbuf, 384, 0, 0, 0, proj_b + l * 384,
                h, 384, 0, 0, 1, nullptr, 0, 0, 0, nullptr, 0,
                T, 384, 384, 1, 1, 0, 1.f);
    // LN2 (fp32 copy only needed for MoE gating)
    k_layernorm<<<T, 128, 0, stream>>>(h, 384, ln2_w + l * 384, ln2_b + l * 384,
                                       lnb, 384, (l & 1) ? lnf32 : nullptr, 384, T, 384);
    if (l & 1) {   // MoE block: top-2 gating, dense all-expert compute, fused combine
      k_moe_gate<<<T, 256, 0, stream>>>(lnf32, gate_w + (long)m * 8 * 384, comb, T, 384, 8);
      for (int e = 0; e < 8; ++e) {
        long wo = (long)(m * 8 + e);
        cvt(e1_w + wo * 1536 * 384, wbuf, 1536L * 384);
        launch_gemm(stream, lnb, 384, 0, 0, wbuf, 384, 0, 0, 0, e1_b + wo * 1536,
                    nullptr, 0, 0, 0, 0, ffnb, 1536, 0, 0, nullptr, 0,
                    T, 1536, 384, 1, 1, 1, 1.f);                       // +GELU
        cvt(e2_w + wo * 384 * 1536, wbuf, 384L * 1536);
        launch_gemm(stream, ffnb, 1536, 0, 0, wbuf, 1536, 0, 0, 0, e2_b + wo * 384,
                    h, 384, 0, 0, 1, nullptr, 0, 0, 0,
                    comb + e, 8,                                       // h += combine[t,e]*(...)
                    T, 384, 1536, 1, 1, 0, 1.f);
      }
    } else {       // dense MLP block
      cvt(fc1_w + (long)m * 1536 * 384, wbuf, 1536L * 384);
      launch_gemm(stream, lnb, 384, 0, 0, wbuf, 384, 0, 0, 0, fc1_b + (long)m * 1536,
                  nullptr, 0, 0, 0, 0, ffnb, 1536, 0, 0, nullptr, 0,
                  T, 1536, 384, 1, 1, 1, 1.f);                         // +GELU
      cvt(fc2_w + (long)m * 384 * 1536, wbuf, 384L * 1536);
      launch_gemm(stream, ffnb, 1536, 0, 0, wbuf, 1536, 0, 0, 0, fc2_b + (long)m * 384,
                  h, 384, 0, 0, 1, nullptr, 0, 0, 0, nullptr, 0,
                  T, 384, 1536, 1, 1, 0, 1.f);
    }
  }

  // final LN on the 4 cls rows only (row stride 197*384), then head GEMM
  k_layernorm<<<4, 128, 0, stream>>>(h, 197L * 384, lnf_w, lnf_b,
                                     clsb, 384, nullptr, 0, 4, 384);
  cvt(head_w, wbuf, 1000L * 384);
  launch_gemm(stream, clsb, 384, 0, 0, wbuf, 384, 0, 0, 0, head_b,
              out, 1000, 0, 0, 0, nullptr, 0, 0, 0, nullptr, 0,
              4, 1000, 384, 1, 1, 0, 1.f);
}
